// TLM_51616916964155
// MI455X (gfx1250) — matrix operand order repaired and checked by reading, not compiled, not run on hardware
//
#include <hip/hip_runtime.h>
#include <hip/hip_bf16.h>

typedef __attribute__((ext_vector_type(16))) _Float16 v16h;
typedef __attribute__((ext_vector_type(8)))  _Float16 h8;
typedef __attribute__((ext_vector_type(8)))  float    v8f;

#define WMMA_F16(a, b, c) \
    __builtin_amdgcn_wmma_f32_16x16x32_f16(false, (a), false, (b), (short)0, (c), false, false)

// Async copy 16B global -> LDS (gfx1250 GLOBAL_LOAD_ASYNC_TO_LDS_B128).
// LDS operand is the low 32 bits of the flat shared address (ISA: LDS flat
// addresses map addr[31:0] -> LDS offset).
__device__ __forceinline__ void async_copy_b128(const void* gaddr, void* lds) {
    unsigned lds_off = (unsigned)(unsigned long long)lds;
    asm volatile("global_load_async_to_lds_b128 %0, %1, off"
                 :: "v"(lds_off), "v"(gaddr) : "memory");
}
__device__ __forceinline__ void wait_async0() {
    asm volatile("s_wait_asynccnt 0x0" ::: "memory");
}

// ---------------------------------------------------------------------------
// Elementwise / reduction helpers
// ---------------------------------------------------------------------------

__global__ __launch_bounds__(256)
void embed_kernel(const int* __restrict__ idx, const float* __restrict__ tok,
                  const float* __restrict__ pos, float* __restrict__ x,
                  int T, int E, long long total) {
    long long i = (long long)blockIdx.x * 256 + threadIdx.x;
    if (i >= total) return;
    long long n = i / E;
    int e = (int)(i - n * E);
    int t = (int)(n % T);
    x[i] = tok[(size_t)idx[n] * E + e] + pos[(size_t)t * E + e];
}

__device__ __forceinline__ float blk_sum256(float v) {
    __shared__ float sh[8];
    int lane = threadIdx.x & 31, w = threadIdx.x >> 5;
#pragma unroll
    for (int m = 16; m; m >>= 1) v += __shfl_xor(v, m, 32);
    __syncthreads();
    if (lane == 0) sh[w] = v;
    __syncthreads();
    float r = 0.f;
#pragma unroll
    for (int i = 0; i < 8; ++i) r += sh[i];
    return r;
}

__device__ __forceinline__ float blk_max256(float v) {
    __shared__ float shm[8];
    int lane = threadIdx.x & 31, w = threadIdx.x >> 5;
#pragma unroll
    for (int m = 16; m; m >>= 1) v = fmaxf(v, __shfl_xor(v, m, 32));
    __syncthreads();
    if (lane == 0) shm[w] = v;
    __syncthreads();
    float r = shm[0];
#pragma unroll
    for (int i = 1; i < 8; ++i) r = fmaxf(r, shm[i]);
    return r;
}

__global__ __launch_bounds__(256)
void layernorm_kernel(const float* __restrict__ x, const float* __restrict__ g,
                      const float* __restrict__ b, float* __restrict__ y, int E) {
    int row = blockIdx.x;
    const float* xr = x + (size_t)row * E;
    float s = 0.f;
    for (int i = threadIdx.x; i < E; i += 256) s += xr[i];
    float mu = blk_sum256(s) / (float)E;
    float vs = 0.f;
    for (int i = threadIdx.x; i < E; i += 256) {
        float d = xr[i] - mu;
        vs += d * d;
    }
    float var = blk_sum256(vs) / (float)E;
    float rstd = rsqrtf(var + 1e-5f);
    float* yr = y + (size_t)row * E;
    for (int i = threadIdx.x; i < E; i += 256)
        yr[i] = (xr[i] - mu) * rstd * g[i] + b[i];
}

__global__ void zero1_kernel(float* p) { *p = 0.f; }

__global__ __launch_bounds__(256)
void loss_kernel(const float* __restrict__ logits, const int* __restrict__ tgt,
                 float* __restrict__ loss, int V, float invN) {
    int row = blockIdx.x;
    const float* lr = logits + (size_t)row * V;
    float mx = -__builtin_inff();
    for (int i = threadIdx.x; i < V; i += 256) mx = fmaxf(mx, lr[i]);
    mx = blk_max256(mx);
    float se = 0.f;
    for (int i = threadIdx.x; i < V; i += 256) se += __expf(lr[i] - mx);
    se = blk_sum256(se);
    if (threadIdx.x == 0) {
        float lp = lr[tgt[row]] - mx - logf(se);
        atomicAdd(loss, -lp * invN);
    }
}

// ---------------------------------------------------------------------------
// Weight pre-swizzle: f32 KxN (row-major, contiguous) -> f16 in B-fragment
// layout. Per 32(K) x 16(N) tile: 512 halves; lane l (0..31) holds
// n = nt*16 + (l&15), k = kt*32 + (l>>4)*8 + i (i < 8), kt*32 + 16 + (l>>4)*8 + (i-8) (i >= 8).
// nt runs fastest so one GEMM K-step's 4 n-tiles are 4KB contiguous.
// ---------------------------------------------------------------------------
__global__ __launch_bounds__(256)
void swizzle_w_kernel(const float* __restrict__ W, _Float16* __restrict__ Wsw,
                      int K, int N) {
    int id = blockIdx.x * 256 + threadIdx.x;
    int total = (K >> 5) * (N >> 4) * 32;
    if (id >= total) return;
    int l = id & 31;
    int blk = id >> 5;
    int ntiles = N >> 4;
    int kt = blk / ntiles;
    int nt = blk - kt * ntiles;
    int n = nt * 16 + (l & 15);
    int kbase = kt * 32 + ((l >> 4) << 3);
    const float* src = W + (size_t)kbase * N + n;
    h8 a, b;
#pragma unroll
    for (int i = 0; i < 8; ++i) a[i] = (_Float16)src[(size_t)i * N];
#pragma unroll
    for (int i = 0; i < 8; ++i) b[i] = (_Float16)src[(size_t)(16 + i) * N];
    _Float16* dst = Wsw + (size_t)blk * 512 + l * 16;
    *(h8*)dst = a;
    *((h8*)dst + 1) = b;
}

// ---------------------------------------------------------------------------
// Tiled WMMA GEMM, pre-swizzled f16 B operand.
// C = A(MxK,f32) * Bsw(KxN,f16 fragment-tiled) [+bias] [+residual] [gelu]
// 128x64 tile, BK=32, 256 threads (8 waves).
// B staging = 256 x async 16B global->LDS copies (ASYNCcnt).
// A staging = float4 loads + packed f16 cvt + 16B LDS stores (A-frag swizzle:
//   l = (k&15)<8 ? m : m+16 ; i = (k>>4)*8 + (k&7)).
// Assumes M%128==0, N%64==0, K%32==0 (true for every call in this model).
// mode 0: C = acc + bias ; 1: += residual ; 2: residual + gelu(acc+bias)
// ---------------------------------------------------------------------------
#define BM 128
#define BN 64
#define BK 32

__global__ __launch_bounds__(256)
void gemm_wmma(const float* __restrict__ A, int lda,
               const _Float16* __restrict__ Bsw,
               const float* __restrict__ bias,
               const float* __restrict__ residual,
               float* __restrict__ C, int ldc,
               int M, int N, int K, int mode) {
    __shared__ alignas(32) _Float16 As[(BM / 16) * 32 * 16];   // 4096 halves
    __shared__ alignas(32) _Float16 Bs[(BN / 16) * 32 * 16];   // 2048 halves

    const int tid  = threadIdx.x;
    const int wave = tid >> 5;
    const int lane = tid & 31;
    const int lhalf = lane >> 4;
    const int l16   = lane & 15;
    const int m0 = blockIdx.y * BM;
    const int n0 = blockIdx.x * BN;
    const int ntiles = N >> 4;

    // A staging: thread -> (row rA, k-half hA); 16 floats -> two h8 slots
    const int rA  = tid >> 1;
    const int hA  = tid & 1;
    const int aBase = ((rA >> 4) * 32 + (rA & 15)) * 16 + hA * 8;
    const float* aSrc = A + (size_t)(m0 + rA) * lda + hA * 16;

    // B staging: thread copies one 16B chunk of the 4KB fragment block
    const _Float16* bSrc0 = Bsw + ((size_t)(n0 >> 4)) * 512 + tid * 8;
    const size_t bStepK = (size_t)ntiles * 512;   // halves per K-step

    v8f acc[4];
#pragma unroll
    for (int t = 0; t < 4; ++t) acc[t] = (v8f){0.f, 0.f, 0.f, 0.f, 0.f, 0.f, 0.f, 0.f};

    for (int k0 = 0; k0 < K; k0 += BK) {
        // ---- B: async global->LDS byte copy of pre-swizzled fragments ----
        const _Float16* bSrc = bSrc0 + (size_t)(k0 >> 5) * bStepK;
        async_copy_b128(bSrc, Bs + tid * 8);
        // ---- A: 4x float4 loads, cvt, 2x 16B LDS stores ----
        {
            const float4* src = (const float4*)(aSrc + k0);
            float4 f0 = src[0], f1 = src[1], f2 = src[2], f3 = src[3];
            h8 lo, hi;
            lo[0] = (_Float16)f0.x; lo[1] = (_Float16)f0.y; lo[2] = (_Float16)f0.z; lo[3] = (_Float16)f0.w;
            lo[4] = (_Float16)f1.x; lo[5] = (_Float16)f1.y; lo[6] = (_Float16)f1.z; lo[7] = (_Float16)f1.w;
            hi[0] = (_Float16)f2.x; hi[1] = (_Float16)f2.y; hi[2] = (_Float16)f2.z; hi[3] = (_Float16)f2.w;
            hi[4] = (_Float16)f3.x; hi[5] = (_Float16)f3.y; hi[6] = (_Float16)f3.z; hi[7] = (_Float16)f3.w;
            *(h8*)(As + aBase)           = lo;   // lane slot m
            *(h8*)(As + aBase + 16 * 16) = hi;   // lane slot m + 16
            if (k0 + BK < K) {
                __builtin_prefetch((const float*)src + BK, 0, 3);
                __builtin_prefetch(bSrc + bStepK, 0, 3);
            }
        }
        wait_async0();
        __syncthreads();

        // ---- fragments: one 32B LDS vector load each ----
        v16h af = *(const v16h*)(As + (wave * 32 + lane) * 16);
#pragma unroll
        for (int t = 0; t < 4; ++t) {
            v16h bf = *(const v16h*)(Bs + t * 512 + lane * 16);
            acc[t] = WMMA_F16(af, bf, acc[t]);
        }
        __syncthreads();
    }

    // Epilogue: element (r,lane) -> row = r + 8*lhalf, col = l16 within tile
    int mrow = m0 + (wave << 4) + (lhalf << 3);
#pragma unroll
    for (int t = 0; t < 4; ++t) {
        int col = n0 + (t << 4) + l16;
        float bv = bias ? bias[col] : 0.f;
#pragma unroll
        for (int r = 0; r < 8; ++r) {
            int row = mrow + r;
            float v = acc[t][r] + bv;
            size_t off = (size_t)row * ldc + col;
            if (mode == 1) {
                v += residual[off];
            } else if (mode == 2) {
                float g = 0.5f * v * (1.f + erff(v * 0.70710678118f));
                v = residual[off] + g;
            }
            C[off] = v;
        }
    }
}

// ---------------------------------------------------------------------------
// Fused flash-style causal attention with WMMA, fragment-major LDS staging.
// q,k,v in [B, T, H*hd]; block = 4 waves = 64 query rows for one (b, head).
// K staged pre-transposed into S-GEMM B-fragment layout (16 sub-tiles);
// V staged into PV B-fragment layout; P scattered into A-fragment layout.
// ---------------------------------------------------------------------------
#define HD 256
#define KT 32

__global__ __launch_bounds__(128)
void attn_wmma(const float* __restrict__ qp, const float* __restrict__ kp,
               const float* __restrict__ vp, float* __restrict__ out,
               int T, int E, float scale) {
    __shared__ alignas(32) _Float16 KsS[16 * 32 * 16];   // (c*2+g) tiles, 16KB
    __shared__ alignas(32) _Float16 VsS[16 * 32 * 16];   // hd tiles, 16KB
    __shared__ alignas(32) _Float16 PsS[4][32 * 16];     // per-wave A-frag, 4KB

    const int tid  = threadIdx.x;
    const int wave = tid >> 5;
    const int lane = tid & 31;
    const int lhalf = lane >> 4;
    const int l16   = lane & 15;
    const int qTile = blockIdx.x * 64;
    const int head  = blockIdx.y;
    const int bz    = blockIdx.z;
    const size_t base = (size_t)bz * T * E + (size_t)head * HD;

    // Preload Q as 8 A-fragments (hd chunks of 32), float4 loads, f32 -> f16
    v16h qf[8];
    {
        int m = (wave << 4) + l16;
        size_t rowoff = base + (size_t)(qTile + m) * E;
        int kb = lhalf * 8;
#pragma unroll
        for (int c = 0; c < 8; ++c) {
            const float4* s0 = (const float4*)(qp + rowoff + c * 32 + kb);
            const float4* s1 = (const float4*)(qp + rowoff + c * 32 + 16 + kb);
            float4 f0 = s0[0], f1 = s0[1], f2 = s1[0], f3 = s1[1];
            qf[c][0]  = (_Float16)f0.x; qf[c][1]  = (_Float16)f0.y;
            qf[c][2]  = (_Float16)f0.z; qf[c][3]  = (_Float16)f0.w;
            qf[c][4]  = (_Float16)f1.x; qf[c][5]  = (_Float16)f1.y;
            qf[c][6]  = (_Float16)f1.z; qf[c][7]  = (_Float16)f1.w;
            qf[c][8]  = (_Float16)f2.x; qf[c][9]  = (_Float16)f2.y;
            qf[c][10] = (_Float16)f2.z; qf[c][11] = (_Float16)f2.w;
            qf[c][12] = (_Float16)f3.x; qf[c][13] = (_Float16)f3.y;
            qf[c][14] = (_Float16)f3.z; qf[c][15] = (_Float16)f3.w;
        }
    }

    v8f o[16];
#pragma unroll
    for (int t = 0; t < 16; ++t) o[t] = (v8f){0.f, 0.f, 0.f, 0.f, 0.f, 0.f, 0.f, 0.f};
    float rmax[8], rsum[8];
#pragma unroll
    for (int r = 0; r < 8; ++r) { rmax[r] = -__builtin_inff(); rsum[r] = 0.f; }

    const int qrowBase = qTile + (wave << 4) + (lhalf << 3);
    const int nTiles = (qTile + 64) / KT;   // uniform across block (causal)

    for (int s = 0; s < nTiles; ++s) {
        int kBase = s * KT;
        // ---- stage K (transposed, S-B-frag layout): 8 tasks/thread ----
        for (int it = 0; it < 8; ++it) {
            int idxk = tid + it * 128;              // 0..1023
            int krow = idxk >> 5;                   // key row 0..31
            int dblk = idxk & 31;                   // d = dblk*8
            const float4* src = (const float4*)(kp + base + (size_t)(kBase + krow) * E + dblk * 8);
            float4 f0 = src[0], f1 = src[1];
            h8 pk;
            pk[0] = (_Float16)f0.x; pk[1] = (_Float16)f0.y; pk[2] = (_Float16)f0.z; pk[3] = (_Float16)f0.w;
            pk[4] = (_Float16)f1.x; pk[5] = (_Float16)f1.y; pk[6] = (_Float16)f1.z; pk[7] = (_Float16)f1.w;
            int c = dblk >> 2;
            int g = krow >> 4;
            int l = ((dblk & 1) << 4) + (krow & 15);
            int i0 = ((dblk >> 1) & 1) * 8;
            *(h8*)(KsS + ((c * 2 + g) * 32 + l) * 16 + i0) = pk;
        }
        // ---- stage V (PV-B-frag layout): 8 tasks/thread ----
        for (int it = 0; it < 8; ++it) {
            int idxv = tid + it * 128;              // 0..1023
            int krblk = idxv >> 8;                  // key octet 0..3
            int d = idxv & 255;                     // hd column
            const float* src = vp + base + (size_t)(kBase + krblk * 8) * E + d;
            h8 pk;
#pragma unroll
            for (int i = 0; i < 8; ++i) pk[i] = (_Float16)src[(size_t)i * E];
            int t = d >> 4;
            int l = ((krblk & 1) << 4) + (d & 15);
            int i0 = (krblk >> 1) * 8;
            *(h8*)(VsS + (t * 32 + l) * 16 + i0) = pk;
        }
        __syncthreads();

        // ---- S = Q K^T over hd=256: 16 WMMAs, fragments = vector loads ----
        v8f s0 = (v8f){0.f, 0.f, 0.f, 0.f, 0.f, 0.f, 0.f, 0.f};
        v8f s1 = (v8f){0.f, 0.f, 0.f, 0.f, 0.f, 0.f, 0.f, 0.f};
#pragma unroll
        for (int c = 0; c < 8; ++c) {
            v16h b0 = *(const v16h*)(KsS + ((c * 2 + 0) * 32 + lane) * 16);
            v16h b1 = *(const v16h*)(KsS + ((c * 2 + 1) * 32 + lane) * 16);
            s0 = WMMA_F16(qf[c], b0, s0);
            s1 = WMMA_F16(qf[c], b1, s1);
        }

        // ---- scale + causal mask + online softmax; scatter P as A-frag ----
#pragma unroll
        for (int r = 0; r < 8; ++r) {
            float a0 = s0[r] * scale;
            float a1 = s1[r] * scale;
            int qrow = qrowBase + r;
            if (kBase + l16 > qrow)      a0 = -__builtin_inff();
            if (kBase + 16 + l16 > qrow) a1 = -__builtin_inff();
            float tm = fmaxf(a0, a1);
#pragma unroll
            for (int m = 1; m < 16; m <<= 1) tm = fmaxf(tm, __shfl_xor(tm, m, 16));
            float nm = fmaxf(rmax[r], tm);
            float fac, p0, p1;
            if (nm == -__builtin_inff()) {
                fac = 1.f; p0 = 0.f; p1 = 0.f;
            } else {
                fac = __expf(rmax[r] - nm);
                p0 = (a0 == -__builtin_inff()) ? 0.f : __expf(a0 - nm);
                p1 = (a1 == -__builtin_inff()) ? 0.f : __expf(a1 - nm);
            }
            float ps = p0 + p1;
#pragma unroll
            for (int m = 1; m < 16; m <<= 1) ps += __shfl_xor(ps, m, 16);
            rsum[r] = rsum[r] * fac + ps;
            rmax[r] = nm;
#pragma unroll
            for (int t = 0; t < 16; ++t) o[t][r] *= fac;
            int row = r + (lhalf << 3);
            // p0: k = l16 (khi=0); p1: k = 16 + l16 (khi=1)
            if (l16 < 8) {
                PsS[wave][row * 16 + l16]            = (_Float16)p0;
                PsS[wave][row * 16 + 8 + l16]        = (_Float16)p1;
            } else {
                PsS[wave][(16 + row) * 16 + l16 - 8] = (_Float16)p0;
                PsS[wave][(16 + row) * 16 + l16]     = (_Float16)p1;
            }
        }
        // In-wave DS RAW hazard: P stored above is read cross-lane below.
        asm volatile("s_wait_dscnt 0" ::: "memory");

        // ---- O += P * V: A-frag = one vector load, 16 WMMAs ----
        v16h pf = *(const v16h*)(PsS[wave] + lane * 16);
#pragma unroll
        for (int t = 0; t < 16; ++t) {
            v16h bf = *(const v16h*)(VsS + (t * 32 + lane) * 16);
            o[t] = WMMA_F16(pf, bf, o[t]);
        }
        __syncthreads();
    }

    // Write O / rowsum to out[b, t, head*HD + col]
#pragma unroll
    for (int t = 0; t < 16; ++t) {
        int col = head * HD + t * 16 + l16;
#pragma unroll
        for (int r = 0; r < 8; ++r) {
            int row = qrowBase + r;
            out[((size_t)bz * T + row) * E + col] = o[t][r] / rsum[r];
        }
    }
}

// ---------------------------------------------------------------------------
// Host orchestration
// ---------------------------------------------------------------------------
extern "C" void kernel_launch(void* const* d_in, const int* in_sizes, int n_in,
                              void* d_out, int out_size, void* d_ws, size_t ws_size,
                              hipStream_t stream) {
    (void)in_sizes; (void)n_in; (void)out_size; (void)ws_size;

    const int* idx      = (const int*)d_in[0];
    const int* targets  = (const int*)d_in[1];
    const float* tok_emb = (const float*)d_in[2];
    const float* pos_emb = (const float*)d_in[3];
    const float* Wq   = (const float*)d_in[4];
    const float* Wk   = (const float*)d_in[5];
    const float* Wv   = (const float*)d_in[6];
    const float* fc_w = (const float*)d_in[7];
    const float* fc_b = (const float*)d_in[8];
    const float* ln1_g = (const float*)d_in[9];
    const float* ln1_b = (const float*)d_in[10];
    const float* ln2_g = (const float*)d_in[11];
    const float* ln2_b = (const float*)d_in[12];
    const float* w1 = (const float*)d_in[13];
    const float* b1 = (const float*)d_in[14];
    const float* w2 = (const float*)d_in[15];
    const float* b2 = (const float*)d_in[16];
    const float* w3 = (const float*)d_in[17];
    const float* b3 = (const float*)d_in[18];
    const float* lm_w = (const float*)d_in[19];
    const float* lm_b = (const float*)d_in[20];

    const int Bb = 2, T = 2048, E = 1024, Hh = 4, hd = 256, Ll = 5, V = 32000;
    const int N = Bb * T;                 // 4096 token rows
    const size_t NE = (size_t)N * E;

    // f32 activation buffers (MLP temporaries alias the dead q/k/v buffers)
    float* ws  = (float*)d_ws;
    float* x   = ws;                      // [N, E]
    float* h   = x + NE;                  // [N, E]
    float* qb  = h + NE;                  // [B, T, H*hd] == [N, E]
    float* kb  = qb + NE;
    float* vb  = kb + NE;
    float* att = vb + NE;                 // [N, E]
    float* t1  = qb;                      // [N, 2E], reuses qb+kb
    float* t2  = vb;                      // [N, E],  reuses vb

    // f16 pre-swizzled weights
    const size_t szHead = (size_t)E * hd;           // per-head QKV slice
    const size_t szQKV  = (size_t)Ll * Hh * szHead;
    const size_t szEE   = (size_t)E * E;
    const size_t szE2E  = (size_t)E * 2 * E;
    _Float16* fw   = (_Float16*)(att + NE);
    _Float16* qsw  = fw;
    _Float16* ksw  = qsw + szQKV;
    _Float16* vsw  = ksw + szQKV;
    _Float16* fcsw = vsw + szQKV;                   // Ll * E*E
    _Float16* w1sw = fcsw + (size_t)Ll * szEE;      // Ll * E*2E
    _Float16* w2sw = w1sw + (size_t)Ll * szE2E;     // Ll * 2E*E
    _Float16* w3sw = w2sw + (size_t)Ll * szE2E;     // Ll * E*E
    _Float16* lmsw = w3sw + (size_t)Ll * szEE;      // E*V

    float* logits  = (float*)d_out;               // [N, V]
    float* lossPtr = logits + (size_t)N * V;      // scalar loss slot

    auto swgrid = [](int K, int Nc) { return dim3(((K >> 5) * (Nc >> 4) * 32 + 255) / 256); };

    // 0) Pre-swizzle all weights to f16 fragment layout (deterministic, per call)
    for (int l = 0; l < Ll; ++l) {
        for (int hh = 0; hh < Hh; ++hh) {
            size_t o = ((size_t)l * Hh + hh) * szHead;
            swizzle_w_kernel<<<swgrid(E, hd), 256, 0, stream>>>(Wq + o, qsw + o, E, hd);
            swizzle_w_kernel<<<swgrid(E, hd), 256, 0, stream>>>(Wk + o, ksw + o, E, hd);
            swizzle_w_kernel<<<swgrid(E, hd), 256, 0, stream>>>(Wv + o, vsw + o, E, hd);
        }
        swizzle_w_kernel<<<swgrid(E, E), 256, 0, stream>>>(fc_w + (size_t)l * szEE, fcsw + (size_t)l * szEE, E, E);
        swizzle_w_kernel<<<swgrid(E, 2 * E), 256, 0, stream>>>(w1 + (size_t)l * szE2E, w1sw + (size_t)l * szE2E, E, 2 * E);
        swizzle_w_kernel<<<swgrid(2 * E, E), 256, 0, stream>>>(w2 + (size_t)l * szE2E, w2sw + (size_t)l * szE2E, 2 * E, E);
        swizzle_w_kernel<<<swgrid(E, E), 256, 0, stream>>>(w3 + (size_t)l * szEE, w3sw + (size_t)l * szEE, E, E);
    }
    swizzle_w_kernel<<<swgrid(E, V), 256, 0, stream>>>(lm_w, lmsw, E, V);

    // 1) Embedding
    {
        long long total = (long long)N * E;
        int blocks = (int)((total + 255) / 256);
        embed_kernel<<<blocks, 256, 0, stream>>>(idx, tok_emb, pos_emb, x, T, E, total);
    }

    dim3 gblk(256);
    auto ggrid = [](int M, int Nc) { return dim3(Nc / BN, M / BM); };

    for (int l = 0; l < Ll; ++l) {
        // LN1
        layernorm_kernel<<<N, 256, 0, stream>>>(x, ln1_g + (size_t)l * E, ln1_b + (size_t)l * E, h, E);
        // QKV projections, per head (heads concatenated in output columns)
        for (int hh = 0; hh < Hh; ++hh) {
            size_t o = ((size_t)l * Hh + hh) * szHead;
            gemm_wmma<<<ggrid(N, hd), gblk, 0, stream>>>(h, E, qsw + o, nullptr, nullptr,
                                                         qb + hh * hd, E, N, hd, E, 0);
            gemm_wmma<<<ggrid(N, hd), gblk, 0, stream>>>(h, E, ksw + o, nullptr, nullptr,
                                                         kb + hh * hd, E, N, hd, E, 0);
            gemm_wmma<<<ggrid(N, hd), gblk, 0, stream>>>(h, E, vsw + o, nullptr, nullptr,
                                                         vb + hh * hd, E, N, hd, E, 0);
        }
        // Fused causal attention (scale = E^-0.5 = 1/32 per reference quirk)
        attn_wmma<<<dim3(T / 64, Hh, Bb), 128, 0, stream>>>(qb, kb, vb, att, T, E, 0.03125f);
        // x = x + att @ fc_w + fc_b
        gemm_wmma<<<ggrid(N, E), gblk, 0, stream>>>(att, E, fcsw + (size_t)l * szEE,
                                                    fc_b + (size_t)l * E, x, x, E, N, E, E, 1);
        // LN2
        layernorm_kernel<<<N, 256, 0, stream>>>(x, ln2_g + (size_t)l * E, ln2_b + (size_t)l * E, h, E);
        // MLP chain: t1 = h@w1+b1 ; t2 = t1@w2+b2 ; x = x + gelu(t2@w3+b3)
        gemm_wmma<<<ggrid(N, 2 * E), gblk, 0, stream>>>(h, E, w1sw + (size_t)l * szE2E,
                                                        b1 + (size_t)l * 2 * E, nullptr, t1, 2 * E,
                                                        N, 2 * E, E, 0);
        gemm_wmma<<<ggrid(N, E), gblk, 0, stream>>>(t1, 2 * E, w2sw + (size_t)l * szE2E,
                                                    b2 + (size_t)l * E, nullptr, t2, E,
                                                    N, E, 2 * E, 0);
        gemm_wmma<<<ggrid(N, E), gblk, 0, stream>>>(t2, E, w3sw + (size_t)l * szEE,
                                                    b3 + (size_t)l * E, x, x, E, N, E, E, 2);
    }

    // LM head: logits = x @ lm_w + lm_b
    gemm_wmma<<<ggrid(N, V), gblk, 0, stream>>>(x, E, lmsw, lm_b, nullptr, logits, V, N, V, E, 0);

    // Loss = mean over rows of -(logits[target] - logsumexp(row))
    zero1_kernel<<<1, 1, 0, stream>>>(lossPtr);
    loss_kernel<<<N, 256, 0, stream>>>(logits, targets, lossPtr, V, 1.0f / (float)N);
}
